// GraphAggregator_12412455485952
// MI455X (gfx1250) — compile-verified, matrix operand order checked by reading, not run on hardware
//
#include <hip/hip_runtime.h>
#include <stdint.h>

// ---------- types ----------
typedef __attribute__((ext_vector_type(16))) __bf16 v16bf;
typedef __attribute__((ext_vector_type(8)))  __bf16 v8bf;
typedef __attribute__((ext_vector_type(4)))  __bf16 v4bf;
typedef __attribute__((ext_vector_type(8)))  float  v8f;

#define DDIM 256
#define GDIM 256
#define SEGS 4096
#define MTILE 128
#define XT_STRIDE 264      // bf16 elements per LDS row (pad 8 -> bank spread)
#define STG_STRIDE 260     // f32 elements per LDS row (pad 4)

// LDS layout (bytes)
#define LDS_XT     0
#define LDS_STG    (MTILE*XT_STRIDE*2)              // 67584
#define LDS_RMAX   (LDS_STG + MTILE*STG_STRIDE*4)   // 200704
#define LDS_RSUM   (LDS_RMAX + MTILE*4)
#define LDS_BS     (LDS_RSUM + MTILE*4)
#define LDS_BGS    (LDS_BS + GDIM*4)
#define LDS_BIDS   (LDS_BGS + GDIM*4)
#define LDS_TOTAL  (LDS_BIDS + MTILE*4)             // 204288 < 320KB WGP LDS

// ---------- WMMA helpers ----------
__device__ __forceinline__ v8f wmma_bf16(v16bf a, v16bf b, v8f c) {
    return __builtin_amdgcn_wmma_f32_16x16x32_bf16(
        /*neg_a=*/false, a, /*neg_b=*/false, b,
        /*c_mod=*/(short)0, c, /*reuse_a=*/false, /*reuse_b=*/false);
}

// Load a 16x32 bf16 fragment from a row-major bf16 matrix.
// Works for both A fragments (rows = M) and B fragments from a
// pre-transposed weight matrix Wt[N][K] (rows = N).
// Per ISA layout: lanes 0-15 take K = kBase+{0..7,16..23},
// lanes 16-31 take K = kBase+{8..15,24..31}, row = base + (lane&15).
__device__ __forceinline__ v16bf load_frag(const __bf16* __restrict__ base,
                                           int rowStride, int row16, int kBase, int lane) {
    int r = row16 + (lane & 15);
    int k = kBase + ((lane >> 4) << 3);
    const __bf16* p = base + r * rowStride + k;
    v8bf lo = *(const v8bf*)(p);
    v8bf hi = *(const v8bf*)(p + 16);
    return __builtin_shufflevector(lo, hi, 0,1,2,3,4,5,6,7,8,9,10,11,12,13,14,15);
}

// ---------- small kernels ----------
extern "C" __global__ void zero_f32(float* p, int n) {
    int i = blockIdx.x * blockDim.x + threadIdx.x;
    if (i < n) p[i] = 0.0f;
}

extern "C" __global__ void prep_weights(const float* __restrict__ W,
                                        const float* __restrict__ Wg,
                                        __bf16* __restrict__ Wt,
                                        __bf16* __restrict__ Wgt) {
    int idx = blockIdx.x * blockDim.x + threadIdx.x;     // idx = d*256 + g
    if (idx < DDIM * GDIM) {
        int d = idx >> 8, g = idx & 255;
        Wt[g * DDIM + d]  = (__bf16)W[idx];
        Wgt[g * DDIM + d] = (__bf16)Wg[idx];
    }
}

extern "C" __global__ void count_kernel(const long long* __restrict__ batch,
                                        float* __restrict__ counts, int N) {
    int i = blockIdx.x * blockDim.x + threadIdx.x;
    if (i < N) atomicAdd(&counts[(int)batch[i]], 1.0f);
}

// ---------- fused GEMM+softmax-gate+segment-scatter ----------
extern "C" __global__ __launch_bounds__(256)
void fused_gate_scatter(const float* __restrict__ x,
                        const long long* __restrict__ batch,
                        const float* __restrict__ bvec,
                        const float* __restrict__ bgvec,
                        const __bf16* __restrict__ Wt,
                        const __bf16* __restrict__ Wgt,
                        float* __restrict__ seg,
                        int N) {
    extern __shared__ char smem[];
    __bf16* xt   = (__bf16*)(smem + LDS_XT);
    float*  stg  = (float*) (smem + LDS_STG);
    float*  rmax = (float*) (smem + LDS_RMAX);
    float*  rsum = (float*) (smem + LDS_RSUM);
    float*  bS   = (float*) (smem + LDS_BS);
    float*  bgS  = (float*) (smem + LDS_BGS);
    int*    bids = (int*)   (smem + LDS_BIDS);

    const int tid  = threadIdx.x;
    const int lane = tid & 31;          // wave32
    const int wave = tid >> 5;          // 8 waves
    const int row0 = blockIdx.x * MTILE;
    const int rBase = wave * 16;        // 16 rows per wave

    if (tid < GDIM) { bS[tid] = bvec[tid]; bgS[tid] = bgvec[tid]; }
    if (tid < MTILE) {
        int i = row0 + tid;
        bids[tid] = (int)batch[i < N ? i : (N - 1)];
    }

    // --- async-copy x tile (f32) into the staging LDS region ---
    {
        unsigned long long xbase = (unsigned long long)(const void*)x;
        for (int idx = tid; idx < MTILE * 64; idx += 256) {   // 64 x 16B chunks/row
            int r = idx >> 6, c = idx & 63;
            if (row0 + r < N) {
                unsigned ldsOff = (unsigned)(unsigned long long)(const void*)
                                  (stg + r * STG_STRIDE + c * 4);
                unsigned long long ga = xbase +
                    ((unsigned long long)(row0 + r) * DDIM + (unsigned)c * 4u) * 4ull;
                asm volatile("global_load_async_to_lds_b128 %0, %1, off"
                             :: "v"(ldsOff), "v"(ga) : "memory");
            }
        }
        asm volatile("s_wait_asynccnt 0" ::: "memory");
    }
    __syncthreads();

    // --- convert f32 tile -> bf16 xt (zero-pad invalid rows) ---
    for (int idx = tid; idx < MTILE * 64; idx += 256) {
        int r = idx >> 6, c4 = (idx & 63) * 4;
        bool ok = (row0 + r) < N;
        v4bf v;
#pragma unroll
        for (int j = 0; j < 4; ++j)
            v[j] = (__bf16)(ok ? stg[r * STG_STRIDE + c4 + j] : 0.0f);
        *(v4bf*)(xt + r * XT_STRIDE + c4) = v;
    }
    __syncthreads();

    // --- pass 1: gate logits = x @ Wg, staged to LDS ---
    for (int nc = 0; nc < 4; ++nc) {
        v8f acc[4] = {};
        const int nb = nc * 64;
#pragma unroll
        for (int ks = 0; ks < 8; ++ks) {
            const int kb = ks * 32;
            v16bf a = load_frag(xt, XT_STRIDE, rBase, kb, lane);
#pragma unroll
            for (int t = 0; t < 4; ++t) {
                v16bf bf_ = load_frag(Wgt, DDIM, nb + t * 16, kb, lane);
                acc[t] = wmma_bf16(a, bf_, acc[t]);
            }
        }
#pragma unroll
        for (int t = 0; t < 4; ++t)
#pragma unroll
            for (int i = 0; i < 8; ++i) {
                int r = rBase + i + ((lane >> 4) << 3);   // C layout: v[i] = rows i / i+8
                int c = nb + t * 16 + (lane & 15);
                stg[r * STG_STRIDE + c] = acc[t][i];
            }
    }
    __syncthreads();

    // --- row softmax stats ---
    if (tid < MTILE) {
        float m = -3.4e38f;
        for (int c = 0; c < GDIM; ++c)
            m = fmaxf(m, stg[tid * STG_STRIDE + c] + bgS[c]);
        float s = 0.0f;
        for (int c = 0; c < GDIM; ++c)
            s += __expf(stg[tid * STG_STRIDE + c] + bgS[c] - m);
        rmax[tid] = m;
        rsum[tid] = 1.0f / s;
    }
    __syncthreads();

    // --- pass 2: states = x @ W, fuse (states+b)*softmax(logit), overwrite stg ---
    for (int nc = 0; nc < 4; ++nc) {
        v8f acc[4] = {};
        const int nb = nc * 64;
#pragma unroll
        for (int ks = 0; ks < 8; ++ks) {
            const int kb = ks * 32;
            v16bf a = load_frag(xt, XT_STRIDE, rBase, kb, lane);
#pragma unroll
            for (int t = 0; t < 4; ++t) {
                v16bf bf_ = load_frag(Wt, DDIM, nb + t * 16, kb, lane);
                acc[t] = wmma_bf16(a, bf_, acc[t]);
            }
        }
#pragma unroll
        for (int t = 0; t < 4; ++t)
#pragma unroll
            for (int i = 0; i < 8; ++i) {
                int r = rBase + i + ((lane >> 4) << 3);
                int c = nb + t * 16 + (lane & 15);
                float logit = stg[r * STG_STRIDE + c] + bgS[c];
                float gate  = __expf(logit - rmax[r]) * rsum[r];
                float val   = (acc[t][i] + bS[c]) * gate;
                if (row0 + r >= N) val = 0.0f;
                stg[r * STG_STRIDE + c] = val;
            }
    }
    __syncthreads();

    // --- segment reduction: batch sorted -> run-length accumulate per column ---
    {
        const int c = tid;              // 256 threads == 256 columns
        int   cur = bids[0];
        float a   = 0.0f;
        for (int r = 0; r < MTILE; ++r) {
            int bid = bids[r];
            float v = stg[r * STG_STRIDE + c];
            if (bid != cur) {
                atomicAdd(&seg[cur * GDIM + c], a);
                cur = bid; a = 0.0f;
            }
            a += v;
        }
        atomicAdd(&seg[cur * GDIM + c], a);
    }
}

// ---------- final tiny GEMM: out = (seg/count) @ Wf + bf ----------
extern "C" __global__ __launch_bounds__(256)
void final_gemm(const float* __restrict__ seg, const float* __restrict__ counts,
                const float* __restrict__ Wf, const float* __restrict__ bf_,
                float* __restrict__ out) {
    __shared__ float row[GDIM];
    int g = blockIdx.x, j = threadIdx.x;
    float inv = 1.0f / fmaxf(counts[g], 1.0f);
    row[j] = seg[g * GDIM + j] * inv;
    __syncthreads();
    float acc = bf_[j];
    for (int k = 0; k < GDIM; ++k)
        acc = fmaf(row[k], Wf[k * GDIM + j], acc);
    out[g * GDIM + j] = acc;
}

// ---------- launcher ----------
extern "C" void kernel_launch(void* const* d_in, const int* in_sizes, int n_in,
                              void* d_out, int out_size, void* d_ws, size_t ws_size,
                              hipStream_t stream) {
    const float*     x     = (const float*)d_in[0];
    const long long* batch = (const long long*)d_in[1];   // jnp.int64
    const float*     W     = (const float*)d_in[2];
    const float*     b     = (const float*)d_in[3];
    const float*     Wg    = (const float*)d_in[4];
    const float*     bg    = (const float*)d_in[5];
    const float*     Wf    = (const float*)d_in[6];
    const float*     bfv   = (const float*)d_in[7];
    float* out = (float*)d_out;
    const int N = in_sizes[1];

    char* ws = (char*)d_ws;
    __bf16* Wt     = (__bf16*)(ws);                          // 128 KB
    __bf16* Wgt    = (__bf16*)(ws + 131072);                 // 128 KB
    float*  seg    = (float*)(ws + 262144);                  // 4 MB
    float*  counts = (float*)(ws + 262144 + SEGS * GDIM * 4);// 16 KB (contig after seg)

    // 1) zero seg + counts (contiguous region) — ws is poisoned by harness
    const int zN = SEGS * GDIM + SEGS;
    zero_f32<<<(zN + 255) / 256, 256, 0, stream>>>(seg, zN);

    // 2) transpose + bf16-convert weights (tiny, L2-resident afterwards)
    prep_weights<<<(DDIM * GDIM + 255) / 256, 256, 0, stream>>>(W, Wg, Wt, Wgt);

    // 3) segment counts
    count_kernel<<<(N + 255) / 256, 256, 0, stream>>>(batch, counts, N);

    // 4) fused GEMM x2 + softmax-gate + segment scatter (bulk of the work)
    const int nblk = (N + MTILE - 1) / MTILE;
    fused_gate_scatter<<<nblk, 256, LDS_TOTAL, stream>>>(x, batch, b, bg, Wt, Wgt, seg, N);

    // 5) mean + final projection
    final_gemm<<<SEGS, 256, 0, stream>>>(seg, counts, Wf, bfv, out);
}
